// VisualGraphAggregator_41111426957463
// MI455X (gfx1250) — matrix operand order repaired and checked by reading, not compiled, not run on hardware
//
#include <hip/hip_runtime.h>
#include <hip/hip_bf16.h>

typedef __attribute__((ext_vector_type(16))) __bf16 v16bf;
typedef __attribute__((ext_vector_type(8)))  __bf16 v8bf;
typedef __attribute__((ext_vector_type(8)))  float  v8f;

// Problem constants (reference shapes)
constexpr int cB    = 128;
constexpr int cNI   = 100;
constexpr int cNS   = 50;
constexpr int cOBJ  = 2048;
constexpr int cSEM  = 300;
constexpr int cPROJ = 1024;
// WMMA-friendly paddings
constexpr int cSEMP = 320;                 // SEM padded to mult of 32
constexpr int cNIP  = 112;                 // NI padded to mult of 16
constexpr int cNSP  = 64;                  // NS padded to mult of 32
constexpr int cMI   = cB * cNI;            // 12800 rows (i tokens)
constexpr int cMS   = cB * 64;             // 8192 padded s rows
constexpr int cCAT  = 2 * cPROJ;           // 2048 concat width

// ---------------- WMMA fragment helpers (CDNA5 wave32 layouts) ----------------
union FragU { v16bf v; v8bf h[2]; };

// A-matrix 16x32 bf16, row-major source [*, lda]:
// lanes 0-15: row=m0+lane, halves K=[k0..k0+7] and [k0+16..k0+23]
// lanes16-31: row=m0+lane-16, halves K=[k0+8..k0+15] and [k0+24..k0+31]
__device__ __forceinline__ v16bf load_fragA(const __bf16* __restrict__ A, int lda, int m0, int k0) {
  int lane = threadIdx.x & 31;
  const __bf16* p = A + (size_t)(m0 + (lane & 15)) * lda + (k0 + ((lane >> 4) << 3));
  FragU f;
  f.h[0] = *(const v8bf*)p;
  f.h[1] = *(const v8bf*)(p + 16);
  return f.v;
}

// B-matrix 32x16 bf16 where B[k][n] = W[n][k], W row-major [N, ldw]:
// lane: n=n0+(lane&15); lanes 0-15 hold K=[k0..k0+7],[k0+16..k0+23], lanes 16-31 K=[k0+8..k0+15],[k0+24..k0+31]
__device__ __forceinline__ v16bf load_fragB(const __bf16* __restrict__ W, int ldw, int n0, int k0) {
  int lane = threadIdx.x & 31;
  const __bf16* p = W + (size_t)(n0 + (lane & 15)) * ldw + (k0 + ((lane >> 4) << 3));
  FragU f;
  f.h[0] = *(const v8bf*)p;
  f.h[1] = *(const v8bf*)(p + 16);
  return f.v;
}

__device__ __forceinline__ v8f wmma_bf16(v16bf a, v16bf b, v8f c) {
  return __builtin_amdgcn_wmma_f32_16x16x32_bf16(false, a, false, b, (short)0, c, false, false);
}

// Wave computes 64(M) x 64(N): 16 WMMA per K-step vs 16 b128 loads (ratio 1.0).
__device__ __forceinline__ void gemm_core_64x64(const __bf16* __restrict__ A, int lda,
                                                const __bf16* __restrict__ W, int ldw,
                                                int m0, int n0, int K, v8f (&acc)[4][4]) {
  for (int k0 = 0; k0 < K; k0 += 32) {
    v16bf a[4];
#pragma unroll
    for (int mi = 0; mi < 4; ++mi) a[mi] = load_fragA(A, lda, m0 + 16 * mi, k0);
#pragma unroll
    for (int j = 0; j < 4; ++j) {
      v16bf bj = load_fragB(W, ldw, n0 + j * 16, k0);
#pragma unroll
      for (int mi = 0; mi < 4; ++mi) acc[mi][j] = wmma_bf16(a[mi], bj, acc[mi][j]);
    }
  }
}

// Wave computes 32(M) x 64(N) (small GEMMs).
__device__ __forceinline__ void gemm_core_32x64(const __bf16* __restrict__ A, int lda,
                                                const __bf16* __restrict__ W, int ldw,
                                                int m0, int n0, int K, v8f (&acc)[2][4]) {
  for (int k0 = 0; k0 < K; k0 += 32) {
    v16bf a0 = load_fragA(A, lda, m0,      k0);
    v16bf a1 = load_fragA(A, lda, m0 + 16, k0);
#pragma unroll
    for (int j = 0; j < 4; ++j) {
      v16bf bj = load_fragB(W, ldw, n0 + j * 16, k0);
      acc[0][j] = wmma_bf16(a0, bj, acc[0][j]);
      acc[1][j] = wmma_bf16(a1, bj, acc[1][j]);
    }
  }
}

// ---------------- Prep kernels ----------------

// weight-norm fold: dst[row][k] = bf16(v[row][k] * g[row] / ||v[row]||), zero pad to Kpad
__global__ __launch_bounds__(256) void prep_weight(const float* __restrict__ v,
                                                   const float* __restrict__ g,
                                                   __bf16* __restrict__ dst,
                                                   int Kreal, int Kpad) {
  int row = blockIdx.x;
  const float* vr = v + (size_t)row * Kreal;
  float ss = 0.f;
  for (int k = threadIdx.x; k < Kreal; k += 256) { float x = vr[k]; ss += x * x; }
#pragma unroll
  for (int off = 16; off > 0; off >>= 1) ss += __shfl_xor(ss, off);
  __shared__ float red[8];
  __shared__ float s_scale;
  if ((threadIdx.x & 31) == 0) red[threadIdx.x >> 5] = ss;
  __syncthreads();
  if (threadIdx.x == 0) {
    float t = 0.f;
#pragma unroll
    for (int wv = 0; wv < 8; ++wv) t += red[wv];
    s_scale = g[row] * rsqrtf(t);
  }
  __syncthreads();
  float scale = s_scale;
  __bf16* dr = dst + (size_t)row * Kpad;
  for (int k = threadIdx.x; k < Kpad; k += 256)
    dr[k] = (k < Kreal) ? (__bf16)(vr[k] * scale) : (__bf16)0.f;
}

// fp32 -> bf16 bulk convert (4 elems/thread)
__global__ __launch_bounds__(256) void f32_to_bf16_v4(const float* __restrict__ in,
                                                      __bf16* __restrict__ out, int n4) {
  int idx = blockIdx.x * 256 + threadIdx.x;
  if (idx >= n4) return;
  const float4 f = ((const float4*)in)[idx];
  typedef __attribute__((ext_vector_type(4))) __bf16 v4bf;
  v4bf o; o[0] = (__bf16)f.x; o[1] = (__bf16)f.y; o[2] = (__bf16)f.z; o[3] = (__bf16)f.w;
  ((v4bf*)out)[idx] = o;
}

// s [B][50][300] fp32 -> s_pad bf16 [B*64][320] (zero padded)
__global__ __launch_bounds__(256) void prep_s_pad(const float* __restrict__ s,
                                                  __bf16* __restrict__ dst) {
  int idx = blockIdx.x * 256 + threadIdx.x;
  int total = cMS * cSEMP;
  if (idx >= total) return;
  int row = idx / cSEMP;
  int k   = idx - row * cSEMP;
  int b = row >> 6;
  int t = row & 63;
  float val = 0.f;
  if (t < cNS && k < cSEM) val = s[((size_t)b * cNS + t) * cSEM + k];
  dst[idx] = (__bf16)val;
}

// softmax over masked pairwise distances -> att bf16 [B*112][64] (zero padded)
__global__ __launch_bounds__(256) void attention_softmax(const float* __restrict__ pi,
                                                         const float* __restrict__ ps,
                                                         const int* __restrict__ mask_s,
                                                         __bf16* __restrict__ att) {
  int gw   = (blockIdx.x * 256 + threadIdx.x) >> 5;  // one wave per (b, m)
  int lane = threadIdx.x & 31;
  int b = gw / cNIP;
  int m = gw - b * cNIP;
  if (b >= cB) return;
  __bf16* orow = att + ((size_t)b * cNIP + m) * cNSP;
  if (m >= cNI) { orow[lane] = (__bf16)0.f; orow[lane + 32] = (__bf16)0.f; return; }
  float px = pi[((size_t)b * cNI + m) * 2 + 0];
  float py = pi[((size_t)b * cNI + m) * 2 + 1];
  int msk = mask_s[b];
  int s0 = lane, s1 = lane + 32;
  bool ok0 = (s0 < cNS) && (s0 < msk);
  bool ok1 = (s1 < cNS) && (s1 < msk);
  float d0 = -INFINITY, d1 = -INFINITY;
  if (ok0) {
    float dx = px - ps[((size_t)b * cNS + s0) * 2 + 0];
    float dy = py - ps[((size_t)b * cNS + s0) * 2 + 1];
    d0 = sqrtf(dx * dx + dy * dy);
  }
  if (ok1) {
    float dx = px - ps[((size_t)b * cNS + s1) * 2 + 0];
    float dy = py - ps[((size_t)b * cNS + s1) * 2 + 1];
    d1 = sqrtf(dx * dx + dy * dy);
  }
  float mx = fmaxf(d0, d1);
#pragma unroll
  for (int off = 16; off > 0; off >>= 1) mx = fmaxf(mx, __shfl_xor(mx, off));
  float e0 = ok0 ? __expf(d0 - mx) : 0.f;
  float e1 = ok1 ? __expf(d1 - mx) : 0.f;
  float sum = e0 + e1;
#pragma unroll
  for (int off = 16; off > 0; off >>= 1) sum += __shfl_xor(sum, off);
  float inv = 1.0f / sum;
  orow[s0] = (__bf16)(e0 * inv);
  orow[s1] = (__bf16)(e1 * inv);
}

// ---------------- WMMA GEMM kernels ----------------

// GEMM1: concat[:, :1024] = relu(i @ wi^T + bi), 64x64 wave tile
__global__ __launch_bounds__(256) void gemm_bias_relu_bf16(
    const __bf16* __restrict__ A, int lda,
    const __bf16* __restrict__ W, int ldw,
    const float* __restrict__ bias,
    __bf16* __restrict__ out, int ldo, int col0, int K) {
  int lane = threadIdx.x & 31;
  int wave = threadIdx.x >> 5;
  int n0 = blockIdx.x * 64;
  int m0 = (blockIdx.y * 8 + wave) * 64;
  v8f acc[4][4] = {};
  gemm_core_64x64(A, lda, W, ldw, m0, n0, K, acc);
  int hi = lane >> 4, lo = lane & 15;
#pragma unroll
  for (int mi = 0; mi < 4; ++mi)
#pragma unroll
    for (int j = 0; j < 4; ++j) {
      int n = n0 + j * 16 + lo;
      float bv = bias[n];
#pragma unroll
      for (int r = 0; r < 8; ++r) {
        int m = m0 + mi * 16 + hi * 8 + r;
        float val = fmaxf(acc[mi][j][r] + bv, 0.f);
        out[(size_t)m * ldo + col0 + n] = (__bf16)val;
      }
    }
}

// GEMM2: s_projT[b][n][t] = relu(s_pad@ws^T + bs)  -- transposed contiguous 16B stores
__global__ __launch_bounds__(256) void gemm_sproj(
    const __bf16* __restrict__ A,   // [B*64 x 320]
    const __bf16* __restrict__ W,   // [1024 x 320]
    const float* __restrict__ bias,
    __bf16* __restrict__ outT) {    // [B][1024][64]
  int lane = threadIdx.x & 31;
  int wave = threadIdx.x >> 5;
  int n0 = blockIdx.x * 64;
  int m0 = (blockIdx.y * 8 + wave) * 32;
  v8f acc[2][4] = {};
  gemm_core_32x64(A, cSEMP, W, cSEMP, m0, n0, cSEMP, acc);
  int hi = lane >> 4, lo = lane & 15;
#pragma unroll
  for (int mi = 0; mi < 2; ++mi) {
    int mrow = m0 + mi * 16 + hi * 8;
    int b = mrow >> 6;
    int t = mrow & 63;
#pragma unroll
    for (int j = 0; j < 4; ++j) {
      int n = n0 + j * 16 + lo;
      float bv = bias[n];
      v8bf pack;
#pragma unroll
      for (int r = 0; r < 8; ++r) pack[r] = (__bf16)fmaxf(acc[mi][j][r] + bv, 0.f);
      *(v8bf*)(outT + ((size_t)b * cPROJ + n) * cNSP + t) = pack;
    }
  }
}

// GEMM3: i_att = att @ s_proj (per-batch), store bf16 into concat[:, 1024:2048]
__global__ __launch_bounds__(256) void gemm_att_apply(
    const __bf16* __restrict__ att,     // [B*112 x 64]
    const __bf16* __restrict__ sprojT,  // [B][1024][64]
    __bf16* __restrict__ out) {         // concat [12800 x 2048]
  int lane = threadIdx.x & 31;
  int wave = threadIdx.x >> 5;
  int n0 = blockIdx.x * 64;
  int m0 = (blockIdx.y * 8 + wave) * 16;     // 16-row tiles never straddle a batch (112 = 7*16)
  int b = m0 / cNIP;
  int ml0 = m0 - b * cNIP;
  const __bf16* W = sprojT + (size_t)b * cPROJ * cNSP;
  v8f acc[4] = {};
  for (int k0 = 0; k0 < cNSP; k0 += 32) {
    v16bf a = load_fragA(att, cNSP, m0, k0);
#pragma unroll
    for (int j = 0; j < 4; ++j)
      acc[j] = wmma_bf16(a, load_fragB(W, cNSP, n0 + j * 16, k0), acc[j]);
  }
  int hi = lane >> 4, lo = lane & 15;
#pragma unroll
  for (int j = 0; j < 4; ++j) {
    int n = n0 + j * 16 + lo;
#pragma unroll
    for (int r = 0; r < 8; ++r) {
      int mloc = ml0 + hi * 8 + r;
      if (mloc < cNI)
        out[((size_t)(b * cNI + mloc)) * cCAT + cPROJ + n] = (__bf16)acc[j][r];
    }
  }
}

// GEMM4: out = i + relu(concat @ wc^T + bc), fp32 output, 64x64 wave tile
__global__ __launch_bounds__(256) void gemm_final(
    const __bf16* __restrict__ A,   // concat [12800 x 2048]
    const __bf16* __restrict__ W,   // wc [2048 x 2048]
    const float* __restrict__ bias,
    const float* __restrict__ resid,
    float* __restrict__ out) {
  int lane = threadIdx.x & 31;
  int wave = threadIdx.x >> 5;
  int n0 = blockIdx.x * 64;
  int m0 = (blockIdx.y * 8 + wave) * 64;
  v8f acc[4][4] = {};
  gemm_core_64x64(A, cCAT, W, cCAT, m0, n0, cCAT, acc);
  int hi = lane >> 4, lo = lane & 15;
#pragma unroll
  for (int mi = 0; mi < 4; ++mi)
#pragma unroll
    for (int j = 0; j < 4; ++j) {
      int n = n0 + j * 16 + lo;
      float bv = bias[n];
#pragma unroll
      for (int r = 0; r < 8; ++r) {
        int m = m0 + mi * 16 + hi * 8 + r;
        size_t idx = (size_t)m * cOBJ + n;
        out[idx] = resid[idx] + fmaxf(acc[mi][j][r] + bv, 0.f);
      }
    }
}

// ---------------- Launch ----------------
extern "C" void kernel_launch(void* const* d_in, const int* in_sizes, int n_in,
                              void* d_out, int out_size, void* d_ws, size_t ws_size,
                              hipStream_t stream) {
  (void)in_sizes; (void)n_in; (void)out_size; (void)ws_size;
  const float* i_in = (const float*)d_in[0];
  const float* s_in = (const float*)d_in[1];
  const float* pi   = (const float*)d_in[2];
  const float* ps   = (const float*)d_in[3];
  const int*   mask = (const int*)d_in[4];
  const float* vi = (const float*)d_in[5];
  const float* gi = (const float*)d_in[6];
  const float* bi = (const float*)d_in[7];
  const float* vs = (const float*)d_in[8];
  const float* gs = (const float*)d_in[9];
  const float* bs = (const float*)d_in[10];
  const float* vc = (const float*)d_in[11];
  const float* gc = (const float*)d_in[12];
  const float* bc = (const float*)d_in[13];
  float* out = (float*)d_out;

  char* wp = (char*)d_ws;
  auto take = [&](size_t bytes) -> char* {
    char* p = wp;
    wp += (bytes + 255) & ~(size_t)255;
    return p;
  };
  __bf16* wi_b    = (__bf16*)take((size_t)cPROJ * cOBJ  * 2);   //  4 MB
  __bf16* ws_b    = (__bf16*)take((size_t)cPROJ * cSEMP * 2);   //  0.6 MB
  __bf16* wc_b    = (__bf16*)take((size_t)cOBJ  * cCAT  * 2);   //  8 MB
  __bf16* i_b     = (__bf16*)take((size_t)cMI   * cOBJ  * 2);   // 50 MB
  __bf16* s_pad   = (__bf16*)take((size_t)cMS   * cSEMP * 2);   //  5 MB
  __bf16* sprojT  = (__bf16*)take((size_t)cB * cPROJ * cNSP * 2); // 16 MB
  __bf16* att     = (__bf16*)take((size_t)cB * cNIP * cNSP * 2);  //  1.75 MB
  __bf16* concat  = (__bf16*)take((size_t)cMI * cCAT * 2);        // 50 MB

  // 1) fold weight-norm into bf16 weights
  prep_weight<<<cPROJ, 256, 0, stream>>>(vi, gi, wi_b, cOBJ, cOBJ);
  prep_weight<<<cPROJ, 256, 0, stream>>>(vs, gs, ws_b, cSEM, cSEMP);
  prep_weight<<<cOBJ,  256, 0, stream>>>(vc, gc, wc_b, cCAT, cCAT);

  // 2) activations to bf16 (+ padding)
  {
    int n4 = (cMI * cOBJ) / 4;
    f32_to_bf16_v4<<<(n4 + 255) / 256, 256, 0, stream>>>(i_in, i_b, n4);
  }
  prep_s_pad<<<(cMS * cSEMP + 255) / 256, 256, 0, stream>>>(s_in, s_pad);

  // 3) attention softmax -> padded bf16 att
  attention_softmax<<<(cB * cNIP) / 8, 256, 0, stream>>>(pi, ps, mask, att);

  // 4) i_proj = relu(i @ wi^T + bi) -> concat[:, :1024]   (M=12800 = 25*8*64)
  {
    dim3 g(cPROJ / 64, cMI / 512);
    gemm_bias_relu_bf16<<<g, 256, 0, stream>>>(i_b, cOBJ, wi_b, cOBJ, bi,
                                               concat, cCAT, 0, cOBJ);
  }
  // 5) s_proj (transposed store)
  {
    dim3 g(cPROJ / 64, cMS / 256);
    gemm_sproj<<<g, 256, 0, stream>>>(s_pad, ws_b, bs, sprojT);
  }
  // 6) i_att = att @ s_proj -> concat[:, 1024:]
  {
    dim3 g(cPROJ / 64, (cB * cNIP) / 128);
    gemm_att_apply<<<g, 256, 0, stream>>>(att, sprojT, concat);
  }
  // 7) out = i + relu(concat @ wc^T + bc)   (M=12800 = 25*8*64)
  {
    dim3 g(cOBJ / 64, cMI / 512);
    gemm_final<<<g, 256, 0, stream>>>(concat, wc_b, bc, i_in, out);
  }
}
